// MultiHeadAttention_21474836480199
// MI455X (gfx1250) — matrix operand order repaired and checked by reading, not compiled, not run on hardware
//
#include <hip/hip_runtime.h>
#include <hip/hip_bf16.h>

// ---------------------------------------------------------------------------
// MI455X (gfx1250, wave32) multi-head attention:
//   1) QKV projection GEMMs (fp32 -> bf16, WMMA f32_16x16x32_bf16)
//   2) flash-attention, double-buffered async LDS staging (ASYNCcnt path)
//   3) output projection GEMM
// Problem: B=4, N=2048, D_IN=1024, H=16, D_HEAD=64
// ---------------------------------------------------------------------------

typedef __bf16 bf16_t;
typedef __attribute__((ext_vector_type(16))) __bf16 v16bf;
typedef __attribute__((ext_vector_type(8)))  float  v8f;
typedef __attribute__((ext_vector_type(4)))  unsigned int v4u;
typedef __attribute__((ext_vector_type(4)))  int v4i;
typedef __attribute__((ext_vector_type(8)))  int v8i;

// Address-space-qualified pointer types for the async global->LDS builtin
// (param types inferred from the round-2 diagnostic: int4* in AS1/AS3).
typedef __attribute__((address_space(1))) v4i g_v4i;
typedef __attribute__((address_space(3))) v4i l_v4i;

#define BATCH   4
#define SEQ     2048
#define DIN     1024
#define NHEAD   16
#define DHEAD   64
#define DMODEL  1024   // NHEAD * DHEAD
#define MROWS   8192   // BATCH * SEQ

static __device__ __forceinline__ v8f wmma_bf16(v16bf a, v16bf b, v8f c) {
  // (neg_a, A, neg_b, B, c_mod, C, reuse_a, reuse_b)
  return __builtin_amdgcn_wmma_f32_16x16x32_bf16(false, a, false, b, (short)0, c,
                                                 false, false);
}

static __device__ __forceinline__ unsigned int pack_bf16(float a, float b) {
  union { __bf16 h[2]; unsigned int u; } x;
  x.h[0] = (__bf16)a;
  x.h[1] = (__bf16)b;
  return x.u;
}

// Async global->LDS 16B copy (GLOBAL_LOAD_ASYNC_TO_LDS_B128, ASYNCcnt).
// Falls back to a synchronous copy if the builtin is unavailable.
#if __has_builtin(__builtin_amdgcn_global_load_async_to_lds_b128)
#define ASYNC_LDS_PATH 1
#else
#define ASYNC_LDS_PATH 0
#endif

static __device__ __forceinline__ void async_cp_b128(void* lds_dst,
                                                     const void* gsrc) {
#if ASYNC_LDS_PATH
  __builtin_amdgcn_global_load_async_to_lds_b128(
      (g_v4i*)(void*)gsrc, (l_v4i*)lds_dst, 0, 0);
#else
  *(v4u*)lds_dst = *(const v4u*)gsrc;
#endif
}

static __device__ __forceinline__ void wait_async_all() {
  asm volatile("s_wait_asynccnt 0" ::: "memory");
}

// A-matrix fragment, 16x32 bf16, row-major source with row stride (elements).
// Lane L holds row M = L&15. lanes 0-15: K = [0..7],[16..23]; lanes 16-31:
// K = [8..15],[24..31]  (ISA 7.12.2 16-bit A layout). Two 16B loads per lane.
static __device__ __forceinline__ v16bf load_a_frag(const bf16_t* base,
                                                    int row_stride, int k0,
                                                    int lane) {
  const int row  = lane & 15;
  const int half = (lane >> 4) & 1;
  union { v16bf v; struct { v4u lo, hi; } u; } f;
  const bf16_t* p = base + (size_t)row * row_stride + k0 + half * 8;
  f.u.lo = *(const v4u*)(p);
  f.u.hi = *(const v4u*)(p + 16);
  return f.v;
}

// B-matrix fragment, 32x16 bf16, from a column-major tile [n][k] with column
// stride (elements). Lane L holds column N = L&15, K = [half*8..+7],
// [16+half*8..+7] (same order as A). Two 16B loads per lane.
static __device__ __forceinline__ v16bf load_b_frag(const bf16_t* base,
                                                    int col_stride, int k0,
                                                    int lane) {
  const int n    = lane & 15;
  const int half = (lane >> 4) & 1;
  union { v16bf v; struct { v4u lo, hi; } u; } f;
  const bf16_t* p = base + (size_t)n * col_stride + k0 + half * 8;
  f.u.lo = *(const v4u*)(p);
  f.u.hi = *(const v4u*)(p + 16);
  return f.v;
}

// Cross-lane reductions over 16-lane halves via ds_swizzle (xor 1,2,4,8 stays
// inside each half of the wave32, matching the C-fragment row grouping).
#define SWZF(v, pat)                                                          \
  __builtin_bit_cast(float, __builtin_amdgcn_ds_swizzle(                      \
                                __builtin_bit_cast(int, (v)), (pat)))

static __device__ __forceinline__ float half16_max(float v) {
  v = fmaxf(v, SWZF(v, 0x041f));
  v = fmaxf(v, SWZF(v, 0x081f));
  v = fmaxf(v, SWZF(v, 0x101f));
  v = fmaxf(v, SWZF(v, 0x201f));
  return v;
}
static __device__ __forceinline__ float half16_sum(float v) {
  v += SWZF(v, 0x041f);
  v += SWZF(v, 0x081f);
  v += SWZF(v, 0x101f);
  v += SWZF(v, 0x201f);
  return v;
}

// ---------------------------------------------------------------------------
// TDM probe (not launched): exercises TENSOR_LOAD_TO_LDS encoding with a
// zeroed (NULL, count=0) descriptor via inline asm, plus s_wait_tensorcnt.
// Compile-only environment: this validates assembler syntax for future use.
// ---------------------------------------------------------------------------
__global__ void tdm_probe_kernel(float* out) {
  __shared__ __align__(128) float tile[1024];
  v4i g0 = {0, 0, 0, 0};
  v8i g1 = {0, 0, 0, 0, 0, 0, 0, 0};
  v4i g2 = {0, 0, 0, 0};
  v4i g3 = {0, 0, 0, 0};
  asm volatile("tensor_load_to_lds %0, %1, %2, %3"
               :
               : "s"(g0), "s"(g1), "s"(g2), "s"(g3)
               : "memory");
  __builtin_amdgcn_s_wait_tensorcnt(0);
  out[threadIdx.x] = tile[threadIdx.x];
}

// ---------------------------------------------------------------------------
// Kernel 1: QKV projection. out = x @ W + b, converted to bf16.
// TRANSPOSE_V = false: out layout (B, H, N, 64)   (for Q and K)
// TRANSPOSE_V = true : out layout (B, H, 64, N)   (for V -> contiguous PV
//                                                  B-fragments, packed stores)
// Block tile 128x128, K-step 32, 8 waves -> each wave 32x64 (2x4 fragments).
// ---------------------------------------------------------------------------
template <bool TRANSPOSE_V>
__global__ __launch_bounds__(256) void qkv_gemm_kernel(
    const float* __restrict__ X,     // [8192][1024] fp32
    const float* __restrict__ W,     // [1024][1024] fp32
    const float* __restrict__ bias,  // [1024] fp32
    bf16_t* __restrict__ out) {
  __shared__ __align__(16) bf16_t sA[128 * 32];  // [m][k]
  __shared__ __align__(16) bf16_t sB[128 * 32];  // [n][k] (transposed W tile)

  const int t    = threadIdx.x;
  const int lane = t & 31;
  const int wv   = t >> 5;
  const int mt   = wv >> 1;  // 0..3
  const int nt   = wv & 1;   // 0..1
  const int mbase = blockIdx.x * 128;
  const int nbase = blockIdx.y * 128;

  v8f acc[2][4];
#pragma unroll
  for (int i = 0; i < 2; ++i)
#pragma unroll
    for (int f = 0; f < 4; ++f) acc[i][f] = {};

  const int a_m  = t >> 1;          // 0..127
  const int a_k  = (t & 1) * 16;    // 0 / 16
  const int b_kp = t >> 4;          // 0..15 (k pair)
  const int b_n  = (t & 15) * 8;    // 0..120

  for (int kb = 0; kb < DIN; kb += 32) {
    // --- stage A tile (fp32 -> bf16) ---
    {
      const float4* xa =
          (const float4*)(X + (size_t)(mbase + a_m) * DIN + kb + a_k);
      float4 f0 = xa[0], f1 = xa[1], f2 = xa[2], f3 = xa[3];
      v4u lo = {pack_bf16(f0.x, f0.y), pack_bf16(f0.z, f0.w),
                pack_bf16(f1.x, f1.y), pack_bf16(f1.z, f1.w)};
      v4u hi = {pack_bf16(f2.x, f2.y), pack_bf16(f2.z, f2.w),
                pack_bf16(f3.x, f3.y), pack_bf16(f3.z, f3.w)};
      v4u* dst = (v4u*)&sA[a_m * 32 + a_k];
      dst[0] = lo;
      dst[1] = hi;
    }
    // --- stage B tile transposed: sB[n][k] ---
    {
      const float4* w0 =
          (const float4*)(W + (size_t)(kb + 2 * b_kp) * DMODEL + nbase + b_n);
      const float4* w1 = (const float4*)((const float*)w0 + DMODEL);
      float4 r00 = w0[0], r01 = w0[1];
      float4 r10 = w1[0], r11 = w1[1];
      unsigned int* sBu = (unsigned int*)sB;
      float c0[8] = {r00.x, r00.y, r00.z, r00.w, r01.x, r01.y, r01.z, r01.w};
      float c1[8] = {r10.x, r10.y, r10.z, r10.w, r11.x, r11.y, r11.z, r11.w};
#pragma unroll
      for (int j = 0; j < 8; ++j)
        sBu[(b_n + j) * 16 + b_kp] = pack_bf16(c0[j], c1[j]);
    }
    // --- prefetch next K-tile into caches (global_prefetch_b8) ---
    if (kb + 32 < DIN) {
      __builtin_prefetch(X + (size_t)(mbase + a_m) * DIN + kb + 32 + a_k, 0, 1);
      __builtin_prefetch(W + (size_t)(kb + 32 + 2 * b_kp) * DMODEL + nbase + b_n,
                         0, 1);
    }
    __syncthreads();

    v16bf af[2], bf_[4];
#pragma unroll
    for (int i = 0; i < 2; ++i)
      af[i] = load_a_frag(sA + (mt * 32 + i * 16) * 32, 32, 0, lane);
#pragma unroll
    for (int f = 0; f < 4; ++f)
      bf_[f] = load_b_frag(sB + (nt * 64 + f * 16) * 32, 32, 0, lane);
#pragma unroll
    for (int i = 0; i < 2; ++i)
#pragma unroll
      for (int f = 0; f < 4; ++f) acc[i][f] = wmma_bf16(af[i], bf_[f], acc[i][f]);
    __syncthreads();
  }

  // --- epilogue: bias + bf16 store in head-major layout ---
  const int half = lane >> 4;
  const int ln   = lane & 15;
#pragma unroll
  for (int f = 0; f < 4; ++f) {
    const int c  = nbase + nt * 64 + f * 16 + ln;
    const float bv = bias[c];
    const int h = c >> 6;
    const int d = c & 63;
#pragma unroll
    for (int i = 0; i < 2; ++i) {
      const int gm0 = mbase + mt * 32 + i * 16 + half * 8;  // multiple of 8
      const int bi  = gm0 >> 11;
      const int row0 = gm0 & 2047;
      if (TRANSPOSE_V) {
        // rows are unit-stride in (B,H,64,N): pack 8 rows -> one b128 store
        bf16_t* dst =
            out + (((size_t)bi * NHEAD + h) * DHEAD + d) * SEQ + row0;
        v4u pk = {pack_bf16(acc[i][f][0] + bv, acc[i][f][1] + bv),
                  pack_bf16(acc[i][f][2] + bv, acc[i][f][3] + bv),
                  pack_bf16(acc[i][f][4] + bv, acc[i][f][5] + bv),
                  pack_bf16(acc[i][f][6] + bv, acc[i][f][7] + bv)};
        *(v4u*)dst = pk;
      } else {
#pragma unroll
        for (int r = 0; r < 8; ++r) {
          out[(((size_t)bi * NHEAD + h) * SEQ + row0 + r) * DHEAD + d] =
              (bf16_t)(acc[i][f][r] + bv);
        }
      }
    }
  }
}

// ---------------------------------------------------------------------------
// Kernel 2: flash attention with double-buffered async LDS staging.
// grid = (N/128, B*H). Each wave owns 16 q rows, full d=64 accumulator.
// Per 64-key tile: K [key][d], V^T [d][key], mask bytes live in LDS buffers;
// tile i+1 is async-prefetched (ASYNCcnt) while tile i feeds 16 WMMAs.
// ---------------------------------------------------------------------------
__global__ __launch_bounds__(256) void attn_kernel(
    const bf16_t* __restrict__ Q,    // (B,H,N,64)
    const bf16_t* __restrict__ K,    // (B,H,N,64)
    const bf16_t* __restrict__ VT,   // (B,H,64,N)
    const unsigned char* __restrict__ mask,  // (B,N,N) bool
    bf16_t* __restrict__ ctx) {      // (B*N, 1024)
  __shared__ __align__(16) bf16_t sK[2][64 * 64];         // [key][d]  2x8KB
  __shared__ __align__(16) bf16_t sVT[2][64 * 64];        // [d][key]  2x8KB
  __shared__ __align__(16) unsigned char sM[2][128 * 64]; //           2x8KB
  __shared__ __align__(16) bf16_t sP[8][16 * 64];         // per wave   16KB

  const int t    = threadIdx.x;
  const int lane = t & 31;
  const int wv   = t >> 5;
  const int half = lane >> 4;
  const int ln   = lane & 15;
  const int bx   = blockIdx.x;      // q block (128 rows)
  const int by   = blockIdx.y;      // b*16 + h
  const int b    = by >> 4;

  // Per-thread staging assignments (each thread moves 3x32B per tile)
  const int v_d   = t >> 2;
  const int v_seg = (t & 3) * 16;
  const int m_qr  = t >> 1;
  const int m_seg = (t & 1) * 32;

  auto issue_stage = [&](int kt, int buf) {
    // K tile: contiguous 8KB
    const char* ksrc = (const char*)(K + ((size_t)by * SEQ + kt) * DHEAD);
    v4u* kdst = (v4u*)sK[buf];
    async_cp_b128(kdst + t, ksrc + (size_t)t * 16);
    async_cp_b128(kdst + t + 256, ksrc + (size_t)(t + 256) * 16);
    // V^T tile: 64 d-rows x 64 keys
    const bf16_t* vsrc = VT + ((size_t)by * DHEAD + v_d) * SEQ + kt + v_seg;
    v4u* vdst = (v4u*)(sVT[buf] + v_d * 64 + v_seg);
    async_cp_b128(vdst, vsrc);
    async_cp_b128(vdst + 1, vsrc + 8);
    // mask tile: 128 q rows x 64 key bytes
    const unsigned char* msrc =
        mask + ((size_t)b * SEQ + bx * 128 + m_qr) * SEQ + kt + m_seg;
    unsigned char* mdst = sM[buf] + m_qr * 64 + m_seg;
    async_cp_b128(mdst, msrc);
    async_cp_b128(mdst + 16, msrc + 16);
  };

  // Q fragments for this wave's 16 rows (d in two 32-chunks), loaded once.
  const bf16_t* Qp = Q + ((size_t)by * SEQ + bx * 128 + wv * 16) * DHEAD;
  v16bf aq0 = load_a_frag(Qp, DHEAD, 0, lane);
  v16bf aq1 = load_a_frag(Qp, DHEAD, 32, lane);

  v8f o[4];
#pragma unroll
  for (int f = 0; f < 4; ++f) o[f] = {};
  float mrow[8], lrow[8];
#pragma unroll
  for (int r = 0; r < 8; ++r) { mrow[r] = -1e30f; lrow[r] = 0.0f; }

  issue_stage(0, 0);

  for (int kt = 0, it = 0; kt < SEQ; kt += 64, ++it) {
    const int buf = it & 1;
    // my async copies done -> barrier makes *everyone's* copies visible, and
    // proves all waves finished reading buf^1 so we can prefetch into it.
    wait_async_all();
    __syncthreads();
    if (kt + 64 < SEQ) issue_stage(kt + 64, buf ^ 1);

    const bf16_t* cK        = sK[buf];
    const bf16_t* cVT       = sVT[buf];
    const unsigned char* cM = sM[buf];

    // --- S = Q K^T : 4 key sub-tiles of 16 ---
    v8f s4[4];
#pragma unroll
    for (int sub = 0; sub < 4; ++sub) {
      s4[sub] = {};
      v16bf b0 = load_b_frag(cK + (sub * 16) * 64, 64, 0, lane);
      v16bf b1 = load_b_frag(cK + (sub * 16) * 64, 64, 32, lane);
      s4[sub] = wmma_bf16(aq0, b0, s4[sub]);
      s4[sub] = wmma_bf16(aq1, b1, s4[sub]);
    }

    // --- mask + scale (1/sqrt(64)) ---
    float p[4][8];
#pragma unroll
    for (int sub = 0; sub < 4; ++sub) {
#pragma unroll
      for (int r = 0; r < 8; ++r) {
        const int row = wv * 16 + half * 8 + r;
        const unsigned char mk = cM[row * 64 + sub * 16 + ln];
        p[sub][r] = mk ? -1e9f : s4[sub][r] * 0.125f;
      }
    }

    // --- online softmax update ---
    float corr[8];
#pragma unroll
    for (int r = 0; r < 8; ++r) {
      float mx = fmaxf(fmaxf(p[0][r], p[1][r]), fmaxf(p[2][r], p[3][r]));
      mx = half16_max(mx);
      const float mnew = fmaxf(mrow[r], mx);
      corr[r] = __expf(mrow[r] - mnew);
      mrow[r] = mnew;
      float sum = 0.0f;
#pragma unroll
      for (int sub = 0; sub < 4; ++sub) {
        p[sub][r] = __expf(p[sub][r] - mnew);
        sum += p[sub][r];
      }
      sum = half16_sum(sum);
      lrow[r] = lrow[r] * corr[r] + sum;
    }
#pragma unroll
    for (int f = 0; f < 4; ++f)
#pragma unroll
      for (int r = 0; r < 8; ++r) o[f][r] *= corr[r];

    // --- P (C layout) -> bf16 in LDS (row-major), then re-read as A frags ---
    bf16_t* pw = &sP[wv][0];
#pragma unroll
    for (int sub = 0; sub < 4; ++sub)
#pragma unroll
      for (int r = 0; r < 8; ++r)
        pw[(half * 8 + r) * 64 + sub * 16 + ln] = (bf16_t)p[sub][r];
    // wave-private LDS region: only our own DS ops must complete
    asm volatile("s_wait_dscnt 0" ::: "memory");

    v16bf ap0 = load_a_frag(pw, 64, 0, lane);
    v16bf ap1 = load_a_frag(pw, 64, 32, lane);
#pragma unroll
    for (int f = 0; f < 4; ++f) {
      v16bf bv0 = load_b_frag(cVT + (f * 16) * 64, 64, 0, lane);
      v16bf bv1 = load_b_frag(cVT + (f * 16) * 64, 64, 32, lane);
      o[f] = wmma_bf16(ap0, bv0, o[f]);
      o[f] = wmma_bf16(ap1, bv1, o[f]);
    }
    // no trailing barrier: next iteration's wait+barrier orders buffer reuse
  }

  // --- epilogue: normalize and store context bf16 (B*N, 1024) ---
  const int h = by & 15;
#pragma unroll
  for (int r = 0; r < 8; ++r) {
    const float inv = 1.0f / lrow[r];
    const int row = bx * 128 + wv * 16 + half * 8 + r;
#pragma unroll
    for (int f = 0; f < 4; ++f) {
      ctx[((size_t)b * SEQ + row) * DMODEL + h * DHEAD + f * 16 + ln] =
          (bf16_t)(o[f][r] * inv);
    }
  }
}

// ---------------------------------------------------------------------------
// Kernel 3: output projection. out(8192x64) = ctx(8192x1024) @ Wo(1024x64).
// Wo staged transposed ([n][k] bf16) in LDS in two 512-K chunks (64KB).
// ---------------------------------------------------------------------------
__global__ __launch_bounds__(256) void out_proj_kernel(
    const bf16_t* __restrict__ ctx,  // (8192, 1024) bf16
    const float* __restrict__ Wo,    // (1024, 64) fp32
    float* __restrict__ out) {       // (8192, 64) fp32
  __shared__ __align__(16) bf16_t sW[64 * 512];  // [n][k-chunk] 64KB

  const int t    = threadIdx.x;
  const int lane = t & 31;
  const int wv   = t >> 5;
  const int half = lane >> 4;
  const int ln   = lane & 15;
  const int mbase = blockIdx.x * 128;

  v8f acc[4];
#pragma unroll
  for (int f = 0; f < 4; ++f) acc[f] = {};

  const bf16_t* cp = ctx + (size_t)(mbase + wv * 16) * DMODEL;

  for (int ko = 0; ko < DIN; ko += 512) {
    // stage Wo chunk transposed: thread t owns k pair (ko + 2t, ko + 2t + 1)
    {
      const float* w0 = Wo + (size_t)(ko + 2 * t) * DHEAD;
      const float* w1 = w0 + DHEAD;
      unsigned int* sWu = (unsigned int*)sW;
#pragma unroll
      for (int n = 0; n < 64; ++n)
        sWu[n * 256 + t] = pack_bf16(w0[n], w1[n]);
    }
    __syncthreads();

    for (int kb = 0; kb < 512; kb += 32) {
      v16bf a = load_a_frag(cp + ko, DMODEL, kb, lane);
#pragma unroll
      for (int f = 0; f < 4; ++f) {
        v16bf bb = load_b_frag(sW + (f * 16) * 512, 512, kb, lane);
        acc[f] = wmma_bf16(a, bb, acc[f]);
      }
    }
    __syncthreads();
  }

#pragma unroll
  for (int f = 0; f < 4; ++f)
#pragma unroll
    for (int r = 0; r < 8; ++r) {
      const int gm = mbase + wv * 16 + half * 8 + r;
      out[(size_t)gm * DHEAD + f * 16 + ln] = acc[f][r];
    }
}

// ---------------------------------------------------------------------------
extern "C" void kernel_launch(void* const* d_in, const int* in_sizes, int n_in,
                              void* d_out, int out_size, void* d_ws,
                              size_t ws_size, hipStream_t stream) {
  (void)in_sizes; (void)n_in; (void)out_size; (void)ws_size;

  const float* x            = (const float*)d_in[0];
  const unsigned char* mask = (const unsigned char*)d_in[1];  // jax bool = 1B
  const float* Wq = (const float*)d_in[2];
  const float* bq = (const float*)d_in[3];
  const float* Wk = (const float*)d_in[4];
  const float* bk = (const float*)d_in[5];
  const float* Wv = (const float*)d_in[6];
  const float* bv = (const float*)d_in[7];
  const float* Wo = (const float*)d_in[8];
  float* out = (float*)d_out;

  // Workspace layout (bf16): Q (B,H,N,64) | K (B,H,N,64) | V^T (B,H,64,N) |
  // context (B*N, 1024). 16 MB each = 64 MB total.
  char* ws = (char*)d_ws;
  bf16_t* qws   = (bf16_t*)(ws);
  bf16_t* kws   = (bf16_t*)(ws + ((size_t)16 << 20));
  bf16_t* vtws  = (bf16_t*)(ws + ((size_t)32 << 20));
  bf16_t* ctxws = (bf16_t*)(ws + ((size_t)48 << 20));

  dim3 gGemm(MROWS / 128, DMODEL / 128);  // 64 x 8
  qkv_gemm_kernel<false><<<gGemm, 256, 0, stream>>>(x, Wq, bq, qws);
  qkv_gemm_kernel<false><<<gGemm, 256, 0, stream>>>(x, Wk, bk, kws);
  qkv_gemm_kernel<true ><<<gGemm, 256, 0, stream>>>(x, Wv, bv, vtws);

  dim3 gAttn(SEQ / 128, BATCH * NHEAD);   // 16 x 64
  attn_kernel<<<gAttn, 256, 0, stream>>>(qws, kws, vtws, mask, ctxws);

  out_proj_kernel<<<MROWS / 128, 256, 0, stream>>>(ctxws, Wo, out);
}
